// BiasedMHA_Bipartite1G_11596411699778
// MI455X (gfx1250) — matrix operand order repaired and checked by reading, not compiled, not run on hardware
//
#include <hip/hip_runtime.h>
#include <hip/hip_bf16.h>
#include <stdint.h>
#include <stddef.h>

// ---------------------------------------------------------------------------
// BiasedMHA for MI455X (gfx1250): bf16 WMMA GEMMs + fused split-K flash attn.
// ---------------------------------------------------------------------------

typedef __bf16 bf16_t;
typedef __attribute__((ext_vector_type(16))) __bf16 v16bf;
typedef __attribute__((ext_vector_type(8)))  __bf16 v8bf;
typedef __attribute__((ext_vector_type(8)))  float   v8f;

#define D_MODEL 512
#define NH      8
#define HD      64
#define LQ      2048
#define LK      8192
#define NSPLIT  4                 // key-dimension splits (flash-decoding style)
#define SPLEN   (LK / NSPLIT)     // 2048 keys per split
#define NQT     (LQ / 16)         // 128 query tiles
#define LARGE_NEG 10000.0f
#define CLAMP_V   20.0f

static __device__ __forceinline__ v16bf comb(v8bf lo, v8bf hi) {
  v16bf r;
#pragma unroll
  for (int i = 0; i < 8; ++i) { r[i] = lo[i]; r[i + 8] = hi[i]; }
  return r;
}

// A-fragment (16x32 MxK, bf16): lane L -> row m0+(L&15);
// halves 0..7 = K in [k0+8*(L>>4), +8), halves 8..15 = same +16.
static __device__ __forceinline__ v16bf load_frag_a(const bf16_t* base, int ld,
                                                    int m0, int k0, int lane) {
  const bf16_t* p = base + (size_t)(m0 + (lane & 15)) * ld + k0 + ((lane >> 4) << 3);
  v8bf lo = *(const v8bf*)p;
  v8bf hi = *(const v8bf*)(p + 16);
  return comb(lo, hi);
}

// B-fragment (32x16 KxN, bf16) where column N is a contiguous row of `base`
// (i.e. base is [N][K] row-major): lane L -> col n0+(L&15);
// halves 0..7 = K in [k0+8*(L>>4), +8), halves 8..15 = same +16.
static __device__ __forceinline__ v16bf load_frag_b(const bf16_t* base, int ld,
                                                    int n0, int k0, int lane) {
  const bf16_t* p = base + (size_t)(n0 + (lane & 15)) * ld + k0 + ((lane >> 4) << 3);
  v8bf lo = *(const v8bf*)p;
  v8bf hi = *(const v8bf*)(p + 16);
  return comb(lo, hi);
}

static __device__ __forceinline__ v8f wmma_bf16(v16bf a, v16bf b, v8f c) {
  // (neg_a, A, neg_b, B, c_mod, C, reuse_a, reuse_b)
  return __builtin_amdgcn_wmma_f32_16x16x32_bf16(false, a, false, b, (short)0, c,
                                                 false, false);
}

// ---------------------------------------------------------------------------
// f32 -> bf16 conversion
// ---------------------------------------------------------------------------
__global__ void k_cvt_bf16(const float* __restrict__ in, bf16_t* __restrict__ out,
                           int n) {
  int i = (blockIdx.x * blockDim.x + threadIdx.x) * 4;
  if (i + 3 < n) {
    float4 v = *(const float4*)(in + i);
    out[i + 0] = (bf16_t)v.x;
    out[i + 1] = (bf16_t)v.y;
    out[i + 2] = (bf16_t)v.z;
    out[i + 3] = (bf16_t)v.w;
  } else {
    for (int j = i; j < n; ++j) out[j] = (bf16_t)in[j];
  }
}

// ---------------------------------------------------------------------------
// C[M,N] = A[M,K] @ W[N,K]^T + bias[N]   (F.linear semantics)
// MODE 0: bf16 out, row-major [M,ldo]
// MODE 1: bf16 out, transposed [N,ldo] (out[n*ldo+m])
// MODE 2: f32 out,  row-major [M,ldo]
// One wave computes a 32x32 tile with 4 accumulators (2x A-reuse, 2x B-reuse).
// ---------------------------------------------------------------------------
template <int MODE>
__global__ void __launch_bounds__(256)
k_gemm_bias(const bf16_t* __restrict__ A, int lda,
            const bf16_t* __restrict__ W, int ldw,
            const float* __restrict__ bias,
            void* __restrict__ outp, int ldo,
            int M, int N, int Kd) {
  const int lane = threadIdx.x & 31;
  const int wave = blockIdx.x * (blockDim.x >> 5) + (threadIdx.x >> 5);
  const int Nt = N >> 5;
  const int Mt = M >> 5;
  if (wave >= Mt * Nt) return;  // uniform per wave: EXEC stays all-ones for WMMA
  const int m0 = (wave / Nt) << 5;
  const int n0 = (wave % Nt) << 5;

  v8f acc00 = {}, acc01 = {}, acc10 = {}, acc11 = {};
  for (int k = 0; k < Kd; k += 32) {
    v16bf a0 = load_frag_a(A, lda, m0,      k, lane);
    v16bf a1 = load_frag_a(A, lda, m0 + 16, k, lane);
    v16bf b0 = load_frag_b(W, ldw, n0,      k, lane);
    v16bf b1 = load_frag_b(W, ldw, n0 + 16, k, lane);
    acc00 = wmma_bf16(a0, b0, acc00);
    acc01 = wmma_bf16(a0, b1, acc01);
    acc10 = wmma_bf16(a1, b0, acc10);
    acc11 = wmma_bf16(a1, b1, acc11);
  }

  const int rbase = (lane >> 4) << 3;  // C layout: lane holds rows rbase..rbase+7
  const int nc = lane & 15;
  const float bv0 = bias[n0 + nc];
  const float bv1 = bias[n0 + 16 + nc];
#pragma unroll
  for (int r = 0; r < 8; ++r) {
    const int mm0 = m0 + rbase + r;
    const int mm1 = mm0 + 16;
    const float c00 = acc00[r] + bv0, c01 = acc01[r] + bv1;
    const float c10 = acc10[r] + bv0, c11 = acc11[r] + bv1;
    if (MODE == 0) {
      bf16_t* o = (bf16_t*)outp;
      o[(size_t)mm0 * ldo + n0 + nc]      = (bf16_t)c00;
      o[(size_t)mm0 * ldo + n0 + 16 + nc] = (bf16_t)c01;
      o[(size_t)mm1 * ldo + n0 + nc]      = (bf16_t)c10;
      o[(size_t)mm1 * ldo + n0 + 16 + nc] = (bf16_t)c11;
    } else if (MODE == 1) {
      bf16_t* o = (bf16_t*)outp;
      o[(size_t)(n0 + nc)      * ldo + mm0] = (bf16_t)c00;
      o[(size_t)(n0 + 16 + nc) * ldo + mm0] = (bf16_t)c01;
      o[(size_t)(n0 + nc)      * ldo + mm1] = (bf16_t)c10;
      o[(size_t)(n0 + 16 + nc) * ldo + mm1] = (bf16_t)c11;
    } else {
      float* o = (float*)outp;
      o[(size_t)mm0 * ldo + n0 + nc]      = c00;
      o[(size_t)mm0 * ldo + n0 + 16 + nc] = c01;
      o[(size_t)mm1 * ldo + n0 + nc]      = c10;
      o[(size_t)mm1 * ldo + n0 + 16 + nc] = c11;
    }
  }
}

// ---------------------------------------------------------------------------
// Fused split-K flash attention (partial pass).
// Grid: (NQT, NSPLIT). Block = 256 threads = 8 waves; block (qt, s) handles
// query tile [16*qt, 16*qt+16) over keys [s*SPLEN, (s+1)*SPLEN); wave w = head.
// The 8 waves of a block read the SAME bias rows => each bias line hits HBM
// once and is re-hit 7x from near caches (whole 64MB bias fits in 192MB L2).
// Writes UNNORMALIZED partial O plus per-row (m, l) statistics to workspace.
// ---------------------------------------------------------------------------
__global__ void __launch_bounds__(256)
k_flash_part(const bf16_t* __restrict__ qp,    // [LQ, D]   projected Q
             const bf16_t* __restrict__ kp,    // [LK, D]   projected K
             const bf16_t* __restrict__ vpT,   // [D, LK]   projected V^T
             const float* __restrict__ bias,   // [LQ, LK]
             const unsigned char* __restrict__ mask,  // [LK] bool
             float* __restrict__ opart,        // [NSPLIT][NQT][NH][16][64]
             float* __restrict__ mpart,        // [NSPLIT][LQ][NH]
             float* __restrict__ lpart) {      // [NSPLIT][LQ][NH]
  __shared__ alignas(16) bf16_t pbuf[8][16 * 32];  // per-wave P staging tile

  const int lane = threadIdx.x & 31;
  const int h = threadIdx.x >> 5;     // head index = wave-in-block
  const int qt = blockIdx.x;
  const int sp = blockIdx.y;
  const int q0 = qt << 4;             // 16 query rows per block
  const int kbeg = sp * SPLEN;
  const int kend = kbeg + SPLEN;

  const bf16_t* qh = qp + h * HD;
  const bf16_t* kh = kp + h * HD;
  const bf16_t* vh = vpT + (size_t)(h * HD) * LK;

  // Q fragments loaded once (d = 64 => two 16x32 A-fragments)
  const v16bf qa0 = load_frag_a(qh, D_MODEL, q0, 0,  lane);
  const v16bf qa1 = load_frag_a(qh, D_MODEL, q0, 32, lane);

  float m_i[8], l_i[8];
#pragma unroll
  for (int r = 0; r < 8; ++r) { m_i[r] = -3.0e38f; l_i[r] = 0.0f; }
  v8f o0 = {}, o1 = {}, o2 = {}, o3 = {};  // 16q x 64d accumulators

  const int rbase = (lane >> 4) << 3;
  const int nc = lane & 15;
  const float scale = 0.125f;  // 1/sqrt(64)

  for (int kk = kbeg; kk < kend; kk += 32) {
    // ---- scores: S[16q x 32k] = q . k^T, two 16x16 tiles, K-depth 64 ----
    v8f s0 = {}, s1 = {};
    {
      v16bf kb00 = load_frag_b(kh, D_MODEL, kk,      0,  lane);
      v16bf kb01 = load_frag_b(kh, D_MODEL, kk,      32, lane);
      v16bf kb10 = load_frag_b(kh, D_MODEL, kk + 16, 0,  lane);
      v16bf kb11 = load_frag_b(kh, D_MODEL, kk + 16, 32, lane);
      s0 = wmma_bf16(qa0, kb00, s0);
      s0 = wmma_bf16(qa1, kb01, s0);
      s1 = wmma_bf16(qa0, kb10, s1);
      s1 = wmma_bf16(qa1, kb11, s1);
    }

    // ---- bias + padding mask + scale ----
    const int c0 = kk + nc, c1 = c0 + 16;
    const float pm0 = mask[c0] ? LARGE_NEG : 0.0f;
    const float pm1 = mask[c1] ? LARGE_NEG : 0.0f;
    float t0[8], t1[8], cm[8];
#pragma unroll
    for (int r = 0; r < 8; ++r) {
      const float* brow = bias + (size_t)(q0 + rbase + r) * LK;
      __builtin_prefetch(brow + c0 + 32, 0, 3);  // next chunk, keep near (WGP)
      t0[r] = s0[r] * scale + brow[c0] - pm0;
      t1[r] = s1[r] * scale + brow[c1] - pm1;
      cm[r] = fmaxf(t0[r], t1[r]);
    }
    // row max across the 16 lanes holding one C-row (half-wave reduction)
#pragma unroll
    for (int mm = 1; mm < 16; mm <<= 1)
#pragma unroll
      for (int r = 0; r < 8; ++r) cm[r] = fmaxf(cm[r], __shfl_xor(cm[r], mm, 32));

    // ---- online softmax update + write P tile (bf16) to LDS ----
    float corr[8];
#pragma unroll
    for (int r = 0; r < 8; ++r) {
      const float mnew = fmaxf(m_i[r], cm[r]);
      corr[r] = __expf(m_i[r] - mnew);
      m_i[r] = mnew;
      const float p0 = __expf(fmaxf(t0[r] - mnew, -CLAMP_V));
      const float p1 = __expf(fmaxf(t1[r] - mnew, -CLAMP_V));
      l_i[r] = l_i[r] * corr[r] + p0 + p1;  // per-lane partial; reduced at end
      bf16_t* pr = &pbuf[h][(rbase + r) * 32];
      pr[nc]      = (bf16_t)p0;
      pr[nc + 16] = (bf16_t)p1;
    }
#pragma unroll
    for (int r = 0; r < 8; ++r) {
      o0[r] *= corr[r]; o1[r] *= corr[r]; o2[r] *= corr[r]; o3[r] *= corr[r];
    }

    // make the cross-lane LDS P-tile visible before reading it back
    asm volatile("s_wait_dscnt 0" ::: "memory");

    // ---- O += P @ V : A = P (from LDS), B = vpT rows (contiguous keys) ----
    v16bf pa  = load_frag_a(&pbuf[h][0], 32, 0, 0, lane);
    v16bf vb0 = load_frag_b(vh, LK, 0,  kk, lane);
    v16bf vb1 = load_frag_b(vh, LK, 16, kk, lane);
    v16bf vb2 = load_frag_b(vh, LK, 32, kk, lane);
    v16bf vb3 = load_frag_b(vh, LK, 48, kk, lane);
    o0 = wmma_bf16(pa, vb0, o0);
    o1 = wmma_bf16(pa, vb1, o1);
    o2 = wmma_bf16(pa, vb2, o2);
    o3 = wmma_bf16(pa, vb3, o3);
  }

  // reduce l over the half-wave; write UNNORMALIZED partials + stats
#pragma unroll
  for (int mm = 1; mm < 16; mm <<= 1)
#pragma unroll
    for (int r = 0; r < 8; ++r) l_i[r] += __shfl_xor(l_i[r], mm, 32);

  float* obase = opart + (size_t)((sp * NQT + qt) * NH + h) * (16 * 64);
#pragma unroll
  for (int r = 0; r < 8; ++r) {
    float* orow = obase + (rbase + r) * 64 + nc;
    orow[0]  = o0[r];
    orow[16] = o1[r];
    orow[32] = o2[r];
    orow[48] = o3[r];
  }
  if (nc == 0) {  // lanes 0 and 16 each own 8 rows' statistics
#pragma unroll
    for (int r = 0; r < 8; ++r) {
      const int row = q0 + rbase + r;
      mpart[(size_t)(sp * LQ + row) * NH + h] = m_i[r];
      lpart[(size_t)(sp * LQ + row) * NH + h] = l_i[r];
    }
  }
}

// ---------------------------------------------------------------------------
// Combine NSPLIT partials: O = sum_y e^{m_y-m*} O_y, l = sum_y e^{m_y-m*} l_y.
// One thread per (q, head, feature) output element -> bf16 attn [LQ, D].
// ---------------------------------------------------------------------------
__global__ void __launch_bounds__(256)
k_flash_combine(const float* __restrict__ opart,  // [NSPLIT][NQT][NH][16][64]
                const float* __restrict__ mpart,  // [NSPLIT][LQ][NH]
                const float* __restrict__ lpart,  // [NSPLIT][LQ][NH]
                bf16_t* __restrict__ attn) {      // [LQ, D]
  const int idx = blockIdx.x * blockDim.x + threadIdx.x;  // LQ*NH*HD total
  const int f = idx & (HD - 1);
  const int qh = idx >> 6;
  const int h = qh & (NH - 1);
  const int q = qh >> 3;

  float mv[NSPLIT];
  float mmax = -3.0e38f;
#pragma unroll
  for (int s = 0; s < NSPLIT; ++s) {
    mv[s] = mpart[(size_t)(s * LQ + q) * NH + h];
    mmax = fmaxf(mmax, mv[s]);
  }
  float osum = 0.0f, lsum = 0.0f;
#pragma unroll
  for (int s = 0; s < NSPLIT; ++s) {
    const float w = __expf(mv[s] - mmax);
    lsum += w * lpart[(size_t)(s * LQ + q) * NH + h];
    osum += w * opart[(size_t)((s * NQT + (q >> 4)) * NH + h) * (16 * 64) +
                      (q & 15) * 64 + f];
  }
  attn[(size_t)q * D_MODEL + h * HD + f] = (bf16_t)(osum / lsum);
}

// ---------------------------------------------------------------------------
// host launcher
// ---------------------------------------------------------------------------
extern "C" void kernel_launch(void* const* d_in, const int* in_sizes, int n_in,
                              void* d_out, int out_size, void* d_ws, size_t ws_size,
                              hipStream_t stream) {
  const float* Q  = (const float*)d_in[0];
  const float* K  = (const float*)d_in[1];
  const float* V  = (const float*)d_in[2];
  const unsigned char* mask = (const unsigned char*)d_in[3];
  const float* bias  = (const float*)d_in[4];
  const float* W_in  = (const float*)d_in[5];
  const float* b_in  = (const float*)d_in[6];
  const float* W_out = (const float*)d_in[7];
  const float* b_out = (const float*)d_in[8];
  float* out = (float*)d_out;

  // workspace carve-up: bf16 region (~42 MB) then f32 partials (~16.5 MB)
  bf16_t* p = (bf16_t*)d_ws;
  bf16_t* qbf    = p; p += (size_t)LQ * D_MODEL;
  bf16_t* kbf    = p; p += (size_t)LK * D_MODEL;
  bf16_t* vbf    = p; p += (size_t)LK * D_MODEL;
  bf16_t* wibf   = p; p += (size_t)3 * D_MODEL * D_MODEL;
  bf16_t* wobf   = p; p += (size_t)D_MODEL * D_MODEL;
  bf16_t* qproj  = p; p += (size_t)LQ * D_MODEL;
  bf16_t* kproj  = p; p += (size_t)LK * D_MODEL;
  bf16_t* vprojT = p; p += (size_t)D_MODEL * LK;
  bf16_t* attnb  = p; p += (size_t)LQ * D_MODEL;
  float* fp = (float*)p;
  float* opart = fp; fp += (size_t)NSPLIT * NQT * NH * 16 * 64;
  float* mpart = fp; fp += (size_t)NSPLIT * LQ * NH;
  float* lpart = fp; fp += (size_t)NSPLIT * LQ * NH;

  auto cvt = [&](const float* src, bf16_t* dst, int n) {
    const int threads = 256;
    const int blocks = (n / 4 + threads - 1) / threads;
    k_cvt_bf16<<<blocks, threads, 0, stream>>>(src, dst, n);
  };
  cvt(Q, qbf, LQ * D_MODEL);
  cvt(K, kbf, LK * D_MODEL);
  cvt(V, vbf, LK * D_MODEL);
  cvt(W_in, wibf, 3 * D_MODEL * D_MODEL);
  cvt(W_out, wobf, D_MODEL * D_MODEL);

  auto gblk = [](int M, int N) { return ((M >> 5) * (N >> 5) + 7) / 8; };

  // q = Q @ Wq^T + bq        -> bf16 [LQ, D]
  k_gemm_bias<0><<<gblk(LQ, D_MODEL), 256, 0, stream>>>(
      qbf, D_MODEL, wibf, D_MODEL, b_in, qproj, D_MODEL, LQ, D_MODEL, D_MODEL);
  // k = K @ Wk^T + bk        -> bf16 [LK, D]
  k_gemm_bias<0><<<gblk(LK, D_MODEL), 256, 0, stream>>>(
      kbf, D_MODEL, wibf + (size_t)D_MODEL * D_MODEL, D_MODEL, b_in + D_MODEL,
      kproj, D_MODEL, LK, D_MODEL, D_MODEL);
  // v = V @ Wv^T + bv        -> bf16 transposed [D, LK] (for P@V B-fragments)
  k_gemm_bias<1><<<gblk(LK, D_MODEL), 256, 0, stream>>>(
      vbf, D_MODEL, wibf + (size_t)2 * D_MODEL * D_MODEL, D_MODEL,
      b_in + 2 * D_MODEL, vprojT, LK, LK, D_MODEL, D_MODEL);

  // fused biased-masked flash attention, split over keys -> f32 partials
  k_flash_part<<<dim3(NQT, NSPLIT), 256, 0, stream>>>(
      qproj, kproj, vprojT, bias, mask, opart, mpart, lpart);

  // combine partials -> bf16 attn [LQ, D]
  k_flash_combine<<<(LQ * NH * HD) / 256, 256, 0, stream>>>(
      opart, mpart, lpart, attnb);

  // out = attn @ W_out^T + b_out -> f32 [LQ, D]
  k_gemm_bias<2><<<gblk(LQ, D_MODEL), 256, 0, stream>>>(
      attnb, D_MODEL, wobf, D_MODEL, b_out, out, D_MODEL, LQ, D_MODEL, D_MODEL);
}
